// istft_43284680409345
// MI455X (gfx1250) — matrix operand order repaired and checked by reading, not compiled, not run on hardware
//
#include <hip/hip_runtime.h>

// ---------------------------------------------------------------------------
// iSTFT = batched GEMM (frames = [real_k | -imag_k] x [magn; phase] + ac)/1024
// fused with overlap-add scatter (hop=512, n_fft=1024 -> overlap factor 2).
// f16 WMMA (v_wmma_f32_16x16x32_f16) with f32 accumulation.
// A tiles moved by the Tensor Data Mover (tensor_load_to_lds + TENSORcnt).
// ---------------------------------------------------------------------------

typedef _Float16 v16h __attribute__((ext_vector_type(16)));
typedef _Float16 v8h  __attribute__((ext_vector_type(8)));
typedef float    v8f  __attribute__((ext_vector_type(8)));
typedef unsigned int v4u __attribute__((ext_vector_type(4)));
typedef int          v4i __attribute__((ext_vector_type(4)));
typedef int          v8i __attribute__((ext_vector_type(8)));

#define N_FFT   1024
#define N_FREQ  512
#define HOP     512
#define T_TOT   3000
#define L_OUT   1536512          // (3000-1)*512 + 1024
#define KF_TOT  1024             // fused reduction dim: 512 (magn) + 512 (phase)
#define MBLK    128
#define NBLK    128
#define KSTEP   32
#define LDSTRIDE 40              // halves per LDS row (80 B: 16-B aligned, conflict-free)

union AFrag { v16h v; v8h h[2]; };

// ---- Tensor DMA descriptor (D#) bitfield builder ---------------------------
__device__ __forceinline__ void set_field(unsigned* d, int lo, int len,
                                          unsigned long long val) {
    int idx = lo >> 5, sh = lo & 31;
    unsigned long long m = (len >= 64) ? ~0ull : ((1ull << len) - 1ull);
    val &= m;
    d[idx] |= (unsigned)(val << sh);
    if (sh + len > 32) d[idx + 1] |= (unsigned)(val >> (32 - sh));
}

// Issue TENSOR_LOAD_TO_LDS for a [rows x cols(halves)] f16 tile.
// global row stride = gstride halves; LDS rows padded to 80 B (40 halves).
__device__ __forceinline__ void tdm_load_a_tile(unsigned long long gaddr,
                                                unsigned lds_addr,
                                                unsigned rows, unsigned cols,
                                                unsigned gstride) {
    unsigned g0[4] = {0, 0, 0, 0};
    unsigned g1[8] = {0, 0, 0, 0, 0, 0, 0, 0};
    // group0
    set_field(g0, 0, 2, 1);            // count = 1 valid descriptor
    set_field(g0, 32, 32, lds_addr);   // LDS byte address
    set_field(g0, 64, 57, gaddr);      // global byte address of tile start
    set_field(g0, 126, 2, 2);          // type = 2 ("image")
    // group1
    set_field(g1, 16, 2, 1);           // data_size = 1 -> 2 bytes
    set_field(g1, 20, 1, 1);           // pad_enable
    set_field(g1, 22, 3, 3);           // pad_interval = 3 -> every 16 dwords (64 B)
    set_field(g1, 25, 7, 3);           // pad_amount = 3 -> 4 dwords (16 B) -> 80 B rows
    set_field(g1, 48, 32, gstride);    // tensor_dim0 (OOB bound; tiles never OOB)
    set_field(g1, 80, 32, rows);       // tensor_dim1
    set_field(g1, 112, 16, cols);      // tile_dim0 (halves per row)
    set_field(g1, 128, 16, rows);      // tile_dim1 (rows)
    set_field(g1, 160, 48, gstride);   // tensor_dim0_stride (halves)
    v4u v0 = {g0[0], g0[1], g0[2], g0[3]};
    v8i v1 = {(int)g1[0], (int)g1[1], (int)g1[2], (int)g1[3],
              (int)g1[4], (int)g1[5], (int)g1[6], (int)g1[7]};
    v4i z4 = {0, 0, 0, 0};
#if __clang_major__ >= 23
    v8i z8 = {0, 0, 0, 0, 0, 0, 0, 0};
    __builtin_amdgcn_tensor_load_to_lds(v0, v1, z4, z4, z8, 0);
#else
    __builtin_amdgcn_tensor_load_to_lds(v0, v1, z4, z4, 0);
#endif
}

// Build A (f16, [1024 rows k][1024 cols f]) = [real_k | -imag_k] once into d_ws.
__global__ void istft_build_a(const float* __restrict__ rk,
                              const float* __restrict__ ik,
                              _Float16* __restrict__ A) {
    int idx = blockIdx.x * blockDim.x + threadIdx.x;   // 1024*1024 elements
    int k = idx >> 10;
    int f = idx & 1023;
    float v = (f < N_FREQ) ? rk[k * N_FREQ + f] : -ik[k * N_FREQ + (f - N_FREQ)];
    A[idx] = (_Float16)v;
}

// Zero the output (harness poisons d_out with 0xAA before timing).
__global__ void istft_zero(float* __restrict__ out, int n) {
    int i = blockIdx.x * blockDim.x + threadIdx.x;
    int stride = gridDim.x * blockDim.x;
    for (; i < n; i += stride) out[i] = 0.0f;
}

// Main fused GEMM + overlap-add scatter.
// grid = (t_tiles=24, k_tiles=8, batch=16), block = 256 (8 wave32)
__global__ __launch_bounds__(256)
void istft_gemm_scatter(const float* __restrict__ magn,
                        const float* __restrict__ phase,
                        const float* __restrict__ ac,
                        const _Float16* __restrict__ A,
                        float* __restrict__ out) {
    __shared__ __align__(16) _Float16 Bs[NBLK * LDSTRIDE];  // [t][f] transposed tile
    __shared__ __align__(16) _Float16 As[MBLK * LDSTRIDE];  // [k][f] TDM-padded tile

    const int t0   = blockIdx.x * NBLK;
    const int k0   = blockIdx.y * MBLK;
    const int b    = blockIdx.z;
    const int tid  = threadIdx.x;
    const int lane = tid & 31;
    const int wave = tid >> 5;
    const int wave_m = wave & 1;   // 2 waves along M (64 rows each)
    const int wave_n = wave >> 1;  // 4 waves along N (32 cols each)
    const int col  = lane & 15;
    const bool hi  = lane >= 16;

    v8f acc[4][2] = {};            // 64x32 per-wave tile: 4 (M) x 2 (N) frags

    const float* magn_b  = magn  + (size_t)b * N_FREQ * T_TOT;
    const float* phase_b = phase + (size_t)b * N_FREQ * T_TOT;

    const unsigned long long a_base =
        (unsigned long long)(uintptr_t)(A + (size_t)k0 * KF_TOT);
    const unsigned as_lds = (unsigned)(uintptr_t)(&As[0]);

    for (int kf0 = 0; kf0 < KF_TOT; kf0 += KSTEP) {
        // k-tiles of 32 never straddle the magn/phase split at 512
        const float* src = (kf0 < N_FREQ)
                         ? (magn_b  + (size_t)kf0 * T_TOT)
                         : (phase_b + (size_t)(kf0 - N_FREQ) * T_TOT);
        __syncthreads();   // previous iteration's LDS reads complete

        // TDM: DMA the [128 x 32] f16 A tile into LDS (overlaps B staging below)
        if (wave == 0) {
            tdm_load_a_tile(a_base + (unsigned long long)kf0 * 2u,
                            as_lds, MBLK, KSTEP, KF_TOT);
        }

        // Stage B tile [32 f][128 t] -> LDS transposed [t][f], f32 -> f16.
        #pragma unroll
        for (int i = 0; i < 16; ++i) {
            int e  = tid + i * 256;          // 0..4095
            int fl = e >> 7;                 // 0..31
            int tl = e & 127;                // 0..127 (consecutive tid -> coalesced)
            int tg = t0 + tl;
            float v = (tg < T_TOT) ? src[(size_t)fl * T_TOT + tg] : 0.0f;
            Bs[tl * LDSTRIDE + fl] = (_Float16)v;
        }

        if (wave == 0) __builtin_amdgcn_s_wait_tensorcnt(0);
        __syncthreads();   // A (TDM) + B (staged) visible to all waves

        // B fragments: lane = column N, lanes>=16 offset K by +8; chunks at K+0 and K+16.
        AFrag bfrag[2];
        #pragma unroll
        for (int nt = 0; nt < 2; ++nt) {
            int n  = wave_n * 32 + nt * 16 + col;
            int kb = hi ? 8 : 0;
            const _Float16* bp = &Bs[n * LDSTRIDE + kb];
            bfrag[nt].h[0] = *(const v8h*)(bp);
            bfrag[nt].h[1] = *(const v8h*)(bp + 16);
        }

        // A fragments from LDS: lane = row M, lanes>=16 offset K by +8;
        // chunks at K+0 and K+16 within the 32-half (padded-80B) row.
        #pragma unroll
        for (int mt = 0; mt < 4; ++mt) {
            int m = wave_m * 64 + mt * 16 + col;       // local row in tile
            const _Float16* ap = &As[m * LDSTRIDE + (hi ? 8 : 0)];
            AFrag af;
            af.h[0] = *(const v8h*)(ap);
            af.h[1] = *(const v8h*)(ap + 16);
            #pragma unroll
            for (int nt = 0; nt < 2; ++nt) {
                acc[mt][nt] = __builtin_amdgcn_wmma_f32_16x16x32_f16(
                    false, af.v, false, bfrag[nt].v,
                    (short)0, acc[mt][nt], false, false);
            }
        }
    }

    // Epilogue: frames = (acc + ac[t]) / 1024, scatter out[b, t*512 + k].
    // C layout: lane<16 -> M=r, lane>=16 -> M=8+r; lane&15 = column.
    const float inv = 1.0f / (float)N_FFT;
    #pragma unroll
    for (int nt = 0; nt < 2; ++nt) {
        int tg = t0 + wave_n * 32 + nt * 16 + col;
        if (tg >= T_TOT) continue;
        float acv = ac[(size_t)b * T_TOT + tg];
        float* outp = out + (size_t)b * L_OUT + (size_t)tg * HOP
                    + k0 + wave_m * 64 + (hi ? 8 : 0);
        #pragma unroll
        for (int mt = 0; mt < 4; ++mt) {
            #pragma unroll
            for (int r = 0; r < 8; ++r) {
                atomicAdd(outp + mt * 16 + r, (acc[mt][nt][r] + acv) * inv);
            }
        }
    }
}

extern "C" void kernel_launch(void* const* d_in, const int* in_sizes, int n_in,
                              void* d_out, int out_size, void* d_ws, size_t ws_size,
                              hipStream_t stream) {
    const float* magn  = (const float*)d_in[0];   // [16,1,512,3000]
    const float* phase = (const float*)d_in[1];   // [16,1,512,3000]
    const float* ac    = (const float*)d_in[2];   // [16,1,1,3000]
    const float* rk    = (const float*)d_in[3];   // [1024,512]
    const float* ik    = (const float*)d_in[4];   // [1024,512]
    float* out = (float*)d_out;                   // [16,1,1,1536512]

    _Float16* A = (_Float16*)d_ws;                // 1024*1024*2 = 2 MB scratch

    // 1) fused kernel matrix [real_k | -imag_k] -> f16
    istft_build_a<<<(1024 * 1024) / 256, 256, 0, stream>>>(rk, ik, A);
    // 2) zero output (atomic accumulation target)
    istft_zero<<<2048, 256, 0, stream>>>(out, out_size);
    // 3) fused GEMM + overlap-add
    dim3 grid((T_TOT + NBLK - 1) / NBLK,   // 24 t-tiles
              N_FFT / MBLK,                // 8 k-tiles
              16);                         // batch
    istft_gemm_scatter<<<grid, 256, 0, stream>>>(magn, phase, ac, A, out);
}
